// EGNNResidueClassifier_61418032333258
// MI455X (gfx1250) — matrix operand order repaired and checked by reading, not compiled, not run on hardware
//
#include <hip/hip_runtime.h>
#include <hip/hip_bf16.h>

typedef __attribute__((ext_vector_type(16))) _Float16 v16h;
typedef __attribute__((ext_vector_type(8)))  _Float16 v8h;
typedef __attribute__((ext_vector_type(8)))  float    v8f;

#define HID 128
#define NNODES 128
#define NGRAPH 16
#define NNTOT 2048        // NGRAPH*NNODES
#define AS 136            // LDS stride (halfs) for 16x128 A tiles (mult of 8)

// ---------------------------------------------------------------------------
// WMMA fragment helpers (C/D layout per CDNA5 ISA 7.12.2; A/B use contiguous k = 16*half + e)
// ---------------------------------------------------------------------------
__device__ __forceinline__ v8f wmma_f16(v16h a, v16h b, v8f c) {
  return __builtin_amdgcn_wmma_f32_16x16x32_f16(false, a, false, b, (short)0, c,
                                                false, false);
}

__device__ __forceinline__ v16h cat8(v8h lo, v8h hi) {
  return __builtin_shufflevector(lo, hi, 0, 1, 2, 3, 4, 5, 6, 7, 8, 9, 10, 11,
                                 12, 13, 14, 15);
}

// A fragment: 16x32 f16 tile at column k0 of row-major [16][stride] halfs.
// lane m=lane&15 holds row m; element e -> K=k0+16*half+e (contiguous).
// Two b128 loads.
__device__ __forceinline__ v16h load_afrag(const _Float16* A, int stride,
                                           int k0, int lane) {
  int m = lane & 15, hf = lane >> 4;
  const _Float16* p = A + m * stride + k0 + hf * 16;
  v8h lo = *(const v8h*)p;
  v8h hi = *(const v8h*)(p + 8);
  return cat8(lo, hi);
}

// B fragment from TRANSPOSED weights Bt[N=128][K=128] (halfs).
// lane n=n0+(lane&15); element e -> K = k0 + 16*half + e.  Two b128 loads.
__device__ __forceinline__ v16h load_bfrag_t(const _Float16* Bt, int n0,
                                             int k0, int lane) {
  int n = n0 + (lane & 15), hf = lane >> 4;
  const _Float16* p = Bt + (size_t)n * HID + k0 + hf * 16;
  v8h lo = *(const v8h*)p;
  v8h hi = *(const v8h*)(p + 8);
  return cat8(lo, hi);
}

// C/D: lane n=n0+(lane&15); element r -> row (lane/16)*8 + r.
__device__ __forceinline__ void store_cfrag(float* C, int stride, int n0,
                                            v8f acc, int lane) {
  int n = n0 + (lane & 15), hf = lane >> 4;
#pragma unroll
  for (int r = 0; r < 8; ++r) C[(hf * 8 + r) * stride + n] = acc[r];
}

// [16x128] = [16x128] @ [128x128] with register-resident B frags.
__device__ __forceinline__ void gemm_preB(const _Float16* Alds,
                                          const v16h Bf[2][4], float* Clds,
                                          int lane, int wave) {
#pragma unroll
  for (int t = 0; t < 2; ++t) {
    v8f acc = {};
#pragma unroll
    for (int kc = 0; kc < 4; ++kc) {
      v16h a = load_afrag(Alds, AS, kc * 32, lane);
      acc = wmma_f16(a, Bf[t][kc], acc);
    }
    store_cfrag(Clds, HID, (wave * 2 + t) * 16, acc, lane);
  }
}

__device__ __forceinline__ float siluf(float x) {
  return x / (1.f + __expf(-x));
}

// ---------------------------------------------------------------------------
// Transpose + convert one 128x128 f32 block to f16 [N][K]
// ---------------------------------------------------------------------------
__global__ void cvt_t_kernel(const float* __restrict__ src,
                             _Float16* __restrict__ dst) {
  int id = blockIdx.x * 256 + threadIdx.x;  // 0..16383
  int k = id >> 7, n = id & 127;
  dst[n * HID + k] = (_Float16)src[k * HID + n];
}

// ---------------------------------------------------------------------------
// h0 = h_in @ emb_in_w + emb_in_b   (K=21, scalar)
// ---------------------------------------------------------------------------
__global__ void embed_kernel(const float* __restrict__ hin,
                             const float* __restrict__ w,
                             const float* __restrict__ bias,
                             float* __restrict__ h, _Float16* __restrict__ h16) {
  int node = blockIdx.x, t = threadIdx.x;
  float acc = bias[t];
#pragma unroll
  for (int k = 0; k < 21; ++k) acc += hin[node * 21 + k] * w[k * HID + t];
  h[node * HID + t] = acc;
  h16[node * HID + t] = (_Float16)acc;
}

// ---------------------------------------------------------------------------
// a = h@W1a ; bterm = h@W1b      (WMMA, per 16-row tile)
// ---------------------------------------------------------------------------
__global__ void node_ab_kernel(const _Float16* __restrict__ h16,
                               const _Float16* __restrict__ w1aT,
                               const _Float16* __restrict__ w1bT,
                               float* __restrict__ aout,
                               float* __restrict__ bout) {
  int rt = blockIdx.x;
  int lane = threadIdx.x & 31, wave = threadIdx.x >> 5;
  const _Float16* Ag = h16 + (size_t)rt * 16 * HID;
#pragma unroll
  for (int t = 0; t < 2; ++t) {
    int n0 = (wave * 2 + t) * 16;
    v8f acc1 = {}, acc2 = {};
#pragma unroll
    for (int kc = 0; kc < 4; ++kc) {
      v16h a = load_afrag(Ag, HID, kc * 32, lane);
      acc1 = wmma_f16(a, load_bfrag_t(w1aT, n0, kc * 32, lane), acc1);
      acc2 = wmma_f16(a, load_bfrag_t(w1bT, n0, kc * 32, lane), acc2);
    }
    store_cfrag(aout + (size_t)rt * 16 * HID, HID, n0, acc1, lane);
    store_cfrag(bout + (size_t)rt * 16 * HID, HID, n0, acc2, lane);
  }
}

// ---------------------------------------------------------------------------
// Fused edge kernel: one workgroup (128 threads = 4 waves) per node i.
// ---------------------------------------------------------------------------
__global__ void edge_kernel(const float* __restrict__ a,
                            const float* __restrict__ bterm,
                            const float* __restrict__ xin,
                            float* __restrict__ xout,
                            _Float16* __restrict__ magg16,
                            const _Float16* __restrict__ w2T,
                            const _Float16* __restrict__ cw1T,
                            const float* __restrict__ wr,
                            const float* __restrict__ w1last,
                            const float* __restrict__ eb1,
                            const float* __restrict__ eb2,
                            const float* __restrict__ attw,
                            const float* __restrict__ attb,
                            const float* __restrict__ cb1,
                            const float* __restrict__ cw2) {
  __shared__ __align__(16) _Float16 A[16 * AS];
  __shared__ float C[16 * HID];
  __shared__ float cds[16][4];   // cd.x, cd.y, cd.z, radial
  __shared__ float att[16];
  __shared__ float phi[16];
  __shared__ float red[16][8];   // row-wise partial reductions

  int node = blockIdx.x;
  int g = node >> 7;
  int t = threadIdx.x;
  int lane = t & 31, wave = t >> 5;
  int rr = t >> 3;               // reduction row 0..15
  int rq = t & 7;                // reduction slot 0..7
  int rc0 = rq << 4;             // reduction column base

  // weight-stationary B fragments (reused across all 8 j-tiles)
  v16h bw2[2][4], bc1[2][4];
#pragma unroll
  for (int tt = 0; tt < 2; ++tt)
#pragma unroll
    for (int kc = 0; kc < 4; ++kc) {
      bw2[tt][kc] = load_bfrag_t(w2T, (wave * 2 + tt) * 16, kc * 32, lane);
      bc1[tt][kc] = load_bfrag_t(cw1T, (wave * 2 + tt) * 16, kc * 32, lane);
    }

  float a_c = a[(size_t)node * HID + t];
  float wr_c = wr[t];
  float bias_c = w1last[t] + eb1[t];
  float eb2_c = eb2[t];
  float attb_s = attb[0];
  // per-thread 16-column slices for the row reductions
  float attw_r[16], cb1_r[16], cw2_r[16];
#pragma unroll
  for (int c = 0; c < 16; ++c) {
    attw_r[c] = attw[rc0 + c];
    cb1_r[c] = cb1[rc0 + c];
    cw2_r[c] = cw2[rc0 + c];
  }
  float x0 = xin[node * 3 + 0], x1 = xin[node * 3 + 1], x2 = xin[node * 3 + 2];
  float maggacc = 0.f;
  float xacc = 0.f;  // used by t<3 (dimension t)

  for (int jb = 0; jb < 8; ++jb) {
    int jbase = g * NNODES + jb * 16;
    if (jb < 7) {  // prefetch next bterm tile (128 lanes x 64B covers 8KB)
      __builtin_prefetch(bterm + (size_t)(jbase + 16 + rr) * HID + rc0, 0, 1);
    }
    if (t < 16) {
      int j = jbase + t;
      float c0 = x0 - xin[j * 3 + 0];
      float c1 = x1 - xin[j * 3 + 1];
      float c2 = x2 - xin[j * 3 + 2];
      cds[t][0] = c0; cds[t][1] = c1; cds[t][2] = c2;
      cds[t][3] = c0 * c0 + c1 * c1 + c2 * c2;
    }
    __syncthreads();
    // A = silu(pre) ; pre = a[i] + b[j] + radial*w_r + bias
#pragma unroll
    for (int r = 0; r < 16; ++r) {
      float pre = a_c + bterm[(size_t)(jbase + r) * HID + t] +
                  cds[r][3] * wr_c + bias_c;
      A[r * AS + t] = (_Float16)siluf(pre);
    }
    __syncthreads();
    gemm_preB(A, bw2, C, lane, wave);            // silu(pre) @ edge_w2
    __syncthreads();
#pragma unroll
    for (int r = 0; r < 16; ++r)
      C[r * HID + t] = siluf(C[r * HID + t] + eb2_c);  // m (pre-gate)
    __syncthreads();
    {                                             // attention logit, parallel
      float d = 0.f;
#pragma unroll
      for (int c = 0; c < 16; ++c) d += C[rr * HID + rc0 + c] * attw_r[c];
      red[rr][rq] = d;
    }
    __syncthreads();
    if (t < 16) {
      float d = attb_s;
#pragma unroll
      for (int q = 0; q < 8; ++q) d += red[t][q];
      att[t] = 1.f / (1.f + __expf(-d));
    }
    __syncthreads();
#pragma unroll
    for (int r = 0; r < 16; ++r) {                // gate, m_agg, A2 = m (f16)
      float m = C[r * HID + t] * att[r];
      A[r * AS + t] = (_Float16)m;
      if (jbase + r != node) maggacc += m;
    }
    __syncthreads();
    gemm_preB(A, bc1, C, lane, wave);            // m @ coord_w1
    __syncthreads();
    {                                             // phi partials, parallel
      float d = 0.f;
#pragma unroll
      for (int c = 0; c < 16; ++c)
        d += siluf(C[rr * HID + rc0 + c] + cb1_r[c]) * cw2_r[c];
      red[rr][rq] = d;
    }
    __syncthreads();
    if (t < 16) {
      float d = 0.f;
#pragma unroll
      for (int q = 0; q < 8; ++q) d += red[t][q];
      phi[t] = d;
    }
    __syncthreads();
    if (t < 3) {
#pragma unroll
      for (int r = 0; r < 16; ++r)
        if (jbase + r != node) xacc += cds[r][t] * phi[r];
    }
    __syncthreads();
  }
  magg16[(size_t)node * HID + t] = (_Float16)maggacc;
  if (t < 3) xout[node * 3 + t] = xin[node * 3 + t] + xacc * (1.f / 127.f);
}

// ---------------------------------------------------------------------------
// Node update: dh = silu(h@nW1a + m_agg@nW1b + nb1) ; h += dh@nW2 + nb2
// ---------------------------------------------------------------------------
__global__ void node_update_kernel(float* __restrict__ h,
                                   _Float16* __restrict__ h16,
                                   const _Float16* __restrict__ magg16,
                                   const _Float16* __restrict__ nw1aT,
                                   const _Float16* __restrict__ nw1bT,
                                   const _Float16* __restrict__ nw2T,
                                   const float* __restrict__ nb1,
                                   const float* __restrict__ nb2) {
  __shared__ float C[16 * HID];
  __shared__ __align__(16) _Float16 Adh[16 * AS];
  int rt = blockIdx.x, t = threadIdx.x;
  int lane = t & 31, wave = t >> 5;
  const _Float16* Ah = h16 + (size_t)rt * 16 * HID;
  const _Float16* Am = magg16 + (size_t)rt * 16 * HID;
#pragma unroll
  for (int tt = 0; tt < 2; ++tt) {
    int n0 = (wave * 2 + tt) * 16;
    v8f acc = {};
#pragma unroll
    for (int kc = 0; kc < 4; ++kc) {
      acc = wmma_f16(load_afrag(Ah, HID, kc * 32, lane),
                     load_bfrag_t(nw1aT, n0, kc * 32, lane), acc);
      acc = wmma_f16(load_afrag(Am, HID, kc * 32, lane),
                     load_bfrag_t(nw1bT, n0, kc * 32, lane), acc);
    }
    store_cfrag(C, HID, n0, acc, lane);
  }
  __syncthreads();
  float b1 = nb1[t];
#pragma unroll
  for (int r = 0; r < 16; ++r)
    Adh[r * AS + t] = (_Float16)siluf(C[r * HID + t] + b1);
  __syncthreads();
#pragma unroll
  for (int tt = 0; tt < 2; ++tt) {
    int n0 = (wave * 2 + tt) * 16;
    v8f acc = {};
#pragma unroll
    for (int kc = 0; kc < 4; ++kc)
      acc = wmma_f16(load_afrag(Adh, AS, kc * 32, lane),
                     load_bfrag_t(nw2T, n0, kc * 32, lane), acc);
    store_cfrag(C, HID, n0, acc, lane);
  }
  __syncthreads();
  float b2 = nb2[t];
#pragma unroll
  for (int r = 0; r < 16; ++r) {
    size_t idx = (size_t)(rt * 16 + r) * HID + t;
    float v = h[idx] + C[r * HID + t] + b2;
    h[idx] = v;
    h16[idx] = (_Float16)v;
  }
}

// ---------------------------------------------------------------------------
// mean-pool (linear => pool before emb_out), emb_out, relu head
// ---------------------------------------------------------------------------
__global__ void pool_head_kernel(const float* __restrict__ h,
                                 const float* __restrict__ ow,
                                 const float* __restrict__ ob,
                                 const float* __restrict__ hw1,
                                 const float* __restrict__ hb1,
                                 const float* __restrict__ hw2,
                                 const float* __restrict__ hb2,
                                 float* __restrict__ out) {
  __shared__ float p[HID], e[HID], r[HID];
  int g = blockIdx.x, t = threadIdx.x;
  float s = 0.f;
  for (int n = 0; n < NNODES; ++n) s += h[(size_t)(g * NNODES + n) * HID + t];
  p[t] = s * (1.f / (float)NNODES);
  __syncthreads();
  float acc = ob[t];
  for (int k = 0; k < HID; ++k) acc += p[k] * ow[k * HID + t];
  e[t] = acc;
  __syncthreads();
  acc = hb1[t];
  for (int k = 0; k < HID; ++k) acc += e[k] * hw1[k * HID + t];
  r[t] = fmaxf(acc, 0.f);
  __syncthreads();
  if (t < 20) {
    float o = hb2[t];
    for (int k = 0; k < HID; ++k) o += r[k] * hw2[k * 20 + t];
    out[g * 20 + t] = o;
  }
}

// ---------------------------------------------------------------------------
extern "C" void kernel_launch(void* const* d_in, const int* in_sizes, int n_in,
                              void* d_out, int out_size, void* d_ws,
                              size_t ws_size, hipStream_t stream) {
  const float* h_in      = (const float*)d_in[0];
  const float* x_in      = (const float*)d_in[1];
  const float* emb_in_w  = (const float*)d_in[2];
  const float* emb_in_b  = (const float*)d_in[3];
  const float* edge_w1   = (const float*)d_in[4];
  const float* edge_b1   = (const float*)d_in[5];
  const float* edge_w2   = (const float*)d_in[6];
  const float* edge_b2   = (const float*)d_in[7];
  const float* att_w     = (const float*)d_in[8];
  const float* att_b     = (const float*)d_in[9];
  const float* node_w1   = (const float*)d_in[10];
  const float* node_b1   = (const float*)d_in[11];
  const float* node_w2   = (const float*)d_in[12];
  const float* node_b2   = (const float*)d_in[13];
  const float* coord_w1  = (const float*)d_in[14];
  const float* coord_b1  = (const float*)d_in[15];
  const float* coord_w2  = (const float*)d_in[16];
  const float* emb_out_w = (const float*)d_in[17];
  const float* emb_out_b = (const float*)d_in[18];
  const float* head_w1   = (const float*)d_in[19];
  const float* head_b1   = (const float*)d_in[20];
  const float* head_w2   = (const float*)d_in[21];
  const float* head_b2   = (const float*)d_in[22];
  (void)in_sizes; (void)n_in; (void)out_size; (void)ws_size;

  // ---- workspace carve-out (256B aligned), ~5 MB total ----
  char* w = (char*)d_ws;
  auto alloc = [&](size_t bytes) -> void* {
    void* p = (void*)w;
    w += (bytes + 255) & ~(size_t)255;
    return p;
  };
  const size_t MAT = (size_t)HID * HID;  // 16384
  float*    h      = (float*)alloc((size_t)NNTOT * HID * 4);
  _Float16* h16    = (_Float16*)alloc((size_t)NNTOT * HID * 2);
  float*    a      = (float*)alloc((size_t)NNTOT * HID * 4);
  float*    bt     = (float*)alloc((size_t)NNTOT * HID * 4);
  _Float16* magg16 = (_Float16*)alloc((size_t)NNTOT * HID * 2);
  float*    xws0   = (float*)alloc((size_t)NNTOT * 3 * 4);
  float*    xws1   = (float*)alloc((size_t)NNTOT * 3 * 4);
  _Float16* ew1aT  = (_Float16*)alloc(4 * MAT * 2);
  _Float16* ew1bT  = (_Float16*)alloc(4 * MAT * 2);
  _Float16* ew2T   = (_Float16*)alloc(4 * MAT * 2);
  _Float16* cw1T   = (_Float16*)alloc(4 * MAT * 2);
  _Float16* nw1aT  = (_Float16*)alloc(4 * MAT * 2);
  _Float16* nw1bT  = (_Float16*)alloc(4 * MAT * 2);
  _Float16* nw2T   = (_Float16*)alloc(4 * MAT * 2);

  // ---- transpose-convert weights to f16 [N][K] ----
  auto cvtT = [&](const float* s, _Float16* d) {
    cvt_t_kernel<<<64, 256, 0, stream>>>(s, d);
  };
  for (int l = 0; l < 4; ++l) {
    cvtT(edge_w1 + (size_t)l * 258 * HID,             ew1aT + l * MAT);
    cvtT(edge_w1 + (size_t)l * 258 * HID + MAT,       ew1bT + l * MAT);
    cvtT(edge_w2 + l * MAT,                           ew2T  + l * MAT);
    cvtT(coord_w1 + l * MAT,                          cw1T  + l * MAT);
    cvtT(node_w1 + (size_t)l * 256 * HID,             nw1aT + l * MAT);
    cvtT(node_w1 + (size_t)l * 256 * HID + MAT,       nw1bT + l * MAT);
    cvtT(node_w2 + l * MAT,                           nw2T  + l * MAT);
  }

  // ---- input embedding ----
  embed_kernel<<<NNTOT, HID, 0, stream>>>(h_in, emb_in_w, emb_in_b, h, h16);

  // ---- EGNN layers ----
  const float* xc = x_in;
  float* xo = xws0;
  for (int l = 0; l < 4; ++l) {
    node_ab_kernel<<<NNTOT / 16, 128, 0, stream>>>(h16, ew1aT + l * MAT,
                                                   ew1bT + l * MAT, a, bt);

    edge_kernel<<<NNTOT, 128, 0, stream>>>(
        a, bt, xc, xo, magg16,
        ew2T + l * MAT, cw1T + l * MAT,
        edge_w1 + (size_t)l * 258 * HID + 256 * HID,   // w_r row
        edge_w1 + (size_t)l * 258 * HID + 257 * HID,   // edge_attr row
        edge_b1 + l * HID, edge_b2 + l * HID,
        att_w + l * HID, att_b + l,
        coord_b1 + l * HID, coord_w2 + l * HID);

    node_update_kernel<<<NNTOT / 16, 128, 0, stream>>>(
        h, h16, magg16, nw1aT + l * MAT, nw1bT + l * MAT, nw2T + l * MAT,
        node_b1 + l * HID, node_b2 + l * HID);

    xc = xo;
    xo = (xo == xws0) ? xws1 : xws0;
  }

  // ---- pool + emb_out + head ----
  pool_head_kernel<<<NGRAPH, HID, 0, stream>>>(h, emb_out_w, emb_out_b,
                                               head_w1, head_b1, head_w2,
                                               head_b2, (float*)d_out);
}
